// ConvGen345_65481071406421
// MI455X (gfx1250) — matrix operand order repaired and checked by reading, not compiled, not run on hardware
//
#include <hip/hip_runtime.h>
#include <hip/hip_bf16.h>

// MI455X (gfx1250) implementation.
// conv2 (16->32, 3x3) and fc1 (1568->64) are mapped to v_wmma_f32_16x16x32_f16.
// Weights are pre-swizzled into per-lane WMMA B-fragment order so the GEMM
// loops issue pure b128 loads + wmma. Wave32 everywhere.

typedef __attribute__((ext_vector_type(16))) _Float16 v16h;
typedef __attribute__((ext_vector_type(8)))  _Float16 v8h;
typedef __attribute__((ext_vector_type(8)))  float    v8f;

#define BATCH 8192

// ---------------- workspace layout (bytes) ----------------
// lifetimes:  pre->q ; conv1: q->h1 ; conv2: h1->c2 ; pool: c2->h2 ;
//             fc1: h2->h3 ; fc2/bn: h3->o->out
#define OFF_H1  0ull                    // 8192*16*196 f16 = 51,380,224
#define OFF_Q   51380224ull             // 8192*784  f32   = 25,690,112
#define OFF_C2  77070336ull             // 8192*196*32 f16 = 102,760,448
#define OFF_H2  0ull                    // 8192*1568 f16   = 25,690,112 (reuses dead h1)
#define OFF_H3  25690112ull             // 8192*64 f32     =  2,097,152
#define OFF_O   27787264ull             // 8192*4  f32     =    131,072
#define OFF_ACC 27918336ull             // 8 f32 (sum, sumsq)
#define OFF_SS  27918400ull             // 8 f32 (scale, shift)
#define OFF_WB  179830784ull            // conv2 B-frags: 5*2*32*16 f16 = 10,240
#define OFF_FW  179841024ull            // fc1  B-frags: 49*4*32*16 f16 = 200,704
// total ws needed ~ 180,041,728 bytes

// ---------------- preprocess: pixel-unshuffle + cosine feature ----------------
__global__ void k_pre(const float* __restrict__ x, float* __restrict__ q, int total) {
  int t = blockIdx.x * blockDim.x + threadIdx.x;
  if (t >= total) return;
  int ox = t % 28; int tt = t / 28; int oy = tt % 28; int img = tt / 28;
  const float* px = x + (size_t)img * 3136 + (size_t)(2 * oy) * 56 + 2 * ox;
  const float kf = 3.14159265358979323846f / 255.0f;
  float c0 = __cosf(px[0]  * kf);
  float c1 = __cosf(px[1]  * kf);
  float c2 = __cosf(px[56] * kf);
  float c3 = __cosf(px[57] * kf);
  q[(size_t)img * 784 + oy * 28 + ox] = (c0 * (1.0f + c1) + c2 * (1.0f + c3)) * 0.25f;
}

// ---------------- conv1 (1->16, 3x3 SAME) + relu + 2x2 maxpool ----------------
__global__ void k_conv1(const float* __restrict__ q, const float* __restrict__ w1,
                        const float* __restrict__ b1, _Float16* __restrict__ h1, int total) {
  int t = blockIdx.x * blockDim.x + threadIdx.x;
  if (t >= total) return;
  int ox = t % 14; int tt = t / 14; int oy = tt % 14; tt /= 14;
  int co = tt % 16; int img = tt / 16;
  float w[9];
#pragma unroll
  for (int i = 0; i < 9; ++i) w[i] = w1[co * 9 + i];
  float bias = b1[co];
  const float* qi = q + (size_t)img * 784;
  float m = 0.0f;                          // relu >= 0 so 0 is a safe identity
#pragma unroll
  for (int dy = 0; dy < 2; ++dy) {
#pragma unroll
    for (int dx = 0; dx < 2; ++dx) {
      int py = 2 * oy + dy, pxc = 2 * ox + dx;
      float s = bias;
#pragma unroll
      for (int ky = 0; ky < 3; ++ky) {
        int y = py + ky - 1;
        if (y < 0 || y > 27) continue;
#pragma unroll
        for (int kx = 0; kx < 3; ++kx) {
          int xx = pxc + kx - 1;
          if (xx < 0 || xx > 27) continue;
          s += qi[y * 28 + xx] * w[ky * 3 + kx];
        }
      }
      s = s > 0.0f ? s : 0.0f;
      m = s > m ? s : m;
    }
  }
  h1[((size_t)img * 16 + co) * 196 + oy * 14 + ox] = (_Float16)m;
}

// ---------- weight pre-swizzle into WMMA B-fragment register order ----------
// B frag order used here (contiguous K, same as A): lanes 0-15 hold K=0..15 (N=lane), lanes 16-31
// hold K=16..31.  frag[( (chunk*NH + nh)*32 + lane )*16 + j] = W[k][n],
// k = chunk*32 + (lane>>4)*16 + j,  n = nh*16 + (lane&15).
__global__ void k_prepw2(const float* __restrict__ w2, _Float16* __restrict__ frag) {
  int tid = threadIdx.x;                   // 5 chunks * 2 nh * 32 lanes = 320
  if (tid >= 320) return;
  int lane = tid & 31;
  int nh   = (tid >> 5) & 1;
  int c    = tid / 64;
  int h    = lane >> 4;
  int n    = nh * 16 + (lane & 15);
#pragma unroll
  for (int j = 0; j < 16; ++j) {
    int k = c * 32 + h * 16 + j;           // k = ci*9 + ky*3 + kx (OIHW flat)
    float v = (k < 144) ? w2[n * 144 + k] : 0.0f;
    frag[tid * 16 + j] = (_Float16)v;
  }
}

__global__ void k_prepfw1(const float* __restrict__ fw1, _Float16* __restrict__ frag) {
  int tid = blockIdx.x * blockDim.x + threadIdx.x;   // 49*4*32 = 6272
  if (tid >= 6272) return;
  int lane = tid & 31;
  int nh   = (tid >> 5) & 3;
  int c    = tid / 128;
  int h    = lane >> 4;
  int n    = nh * 16 + (lane & 15);
#pragma unroll
  for (int j = 0; j < 16; ++j) {
    int k = c * 32 + h * 16 + j;                     // 1568 = 49*32, no padding
    frag[tid * 16 + j] = (_Float16)fw1[n * 1568 + k];
  }
}

// --------- conv2 as implicit GEMM: M=8192*196, N=32, K=144(->160), WMMA ---------
__global__ void __launch_bounds__(256) k_conv2(const _Float16* __restrict__ h1,
                                               const _Float16* __restrict__ wfrag,
                                               const float* __restrict__ b2,
                                               _Float16* __restrict__ c2) {
  __shared__ __align__(16) _Float16 smem[8 * 16 * 160];  // 40 KB: per-wave im2col panel
  int lane = threadIdx.x & 31;
  int wave = threadIdx.x >> 5;
  long tile  = (long)blockIdx.x * 8 + wave;   // 100,352 tiles total, exact
  long mbase = tile * 16;
  _Float16* my = smem + wave * 2560;

  // cooperative im2col fill: 16 rows x 160 K (K>=144 zero-padded)
  for (int r = 0; r < 16; ++r) {
    long mg = mbase + r;
    int img = (int)(mg / 196);
    int p   = (int)(mg - (long)img * 196);
    int oy  = p / 14, ox = p - oy * 14;
    const _Float16* hi = h1 + (size_t)img * 3136;      // 16*196
    for (int k = lane; k < 160; k += 32) {             // 5 iterations, uniform
      _Float16 v = (_Float16)0.0f;
      if (k < 144) {
        int ci = k / 9;
        int tq = k - ci * 9;
        int ky = tq / 3;
        int kx = tq - ky * 3;
        int iy = oy + ky - 1, ix = ox + kx - 1;
        if (iy >= 0 && iy < 14 && ix >= 0 && ix < 14)
          v = hi[ci * 196 + iy * 14 + ix];
      }
      my[r * 160 + k] = v;
    }
  }
  __syncthreads();

  int h  = lane >> 4;
  int nl = lane & 15;
  v8f acc0 = {};
  v8f acc1 = {};
  const _Float16* arow = my + nl * 160 + h * 16;       // A layout: half h holds K = 16h..16h+15
#pragma unroll
  for (int c = 0; c < 5; ++c) {
    v8h lo  = *(const v8h*)(arow + c * 32);            // K = c*32 + h*16 .. +7
    v8h hi8 = *(const v8h*)(arow + c * 32 + 8);        // K = c*32+h*16+8 .. +15
    v16h a  = __builtin_shufflevector(lo, hi8, 0, 1, 2, 3, 4, 5, 6, 7,
                                      8, 9, 10, 11, 12, 13, 14, 15);
    v16h b0 = *(const v16h*)(wfrag + ((c * 2 + 0) * 32 + lane) * 16);
    v16h b1 = *(const v16h*)(wfrag + ((c * 2 + 1) * 32 + lane) * 16);
    acc0 = __builtin_amdgcn_wmma_f32_16x16x32_f16(false, a, false, b0, (short)0, acc0, false, false);
    acc1 = __builtin_amdgcn_wmma_f32_16x16x32_f16(false, a, false, b1, (short)0, acc1, false, false);
  }

  float bias0 = b2[nl];
  float bias1 = b2[16 + nl];
#pragma unroll
  for (int r = 0; r < 8; ++r) {                        // D: VGPR r -> M = r + 8*h
    long mg = mbase + r + 8 * h;
    float v0 = acc0[r] + bias0; v0 = v0 > 0.0f ? v0 : 0.0f;
    float v1 = acc1[r] + bias1; v1 = v1 > 0.0f ? v1 : 0.0f;
    c2[mg * 32 + nl]      = (_Float16)v0;
    c2[mg * 32 + 16 + nl] = (_Float16)v1;
  }
}

// ---------------- 2x2 maxpool + channel-major flatten to (B,1568) ----------------
__global__ void k_pool(const _Float16* __restrict__ c2, _Float16* __restrict__ h2, int total) {
  int t = blockIdx.x * blockDim.x + threadIdx.x;
  if (t >= total) return;
  int ox = t % 7; int tt = t / 7; int oy = tt % 7; tt /= 7;
  int ci = tt % 32; int img = tt / 32;
  long base = (long)img * 196;
  int p00 = (2 * oy) * 14 + 2 * ox;
  float a = (float)c2[(base + p00)      * 32 + ci];
  float b = (float)c2[(base + p00 + 1)  * 32 + ci];
  float c = (float)c2[(base + p00 + 14) * 32 + ci];
  float d = (float)c2[(base + p00 + 15) * 32 + ci];
  float m = fmaxf(fmaxf(a, b), fmaxf(c, d));
  h2[(long)img * 1568 + ci * 49 + oy * 7 + ox] = (_Float16)m;
}

// ---------------- fc1: (8192,1568)x(1568,64) + relu, WMMA ----------------
__global__ void __launch_bounds__(256) k_fc1(const _Float16* __restrict__ h2,
                                             const _Float16* __restrict__ wfrag,
                                             const float* __restrict__ fb1,
                                             float* __restrict__ h3) {
  int lane = threadIdx.x & 31;
  int wave = threadIdx.x >> 5;
  int tile = blockIdx.x * 8 + wave;                    // 512 tiles, exact
  int h  = lane >> 4;
  int nl = lane & 15;
  const _Float16* arow = h2 + (long)(tile * 16 + nl) * 1568 + h * 16;
  v8f acc[4] = {{}, {}, {}, {}};
  for (int c = 0; c < 49; ++c) {
    v8h lo  = *(const v8h*)(arow + c * 32);
    v8h hi8 = *(const v8h*)(arow + c * 32 + 8);
    v16h a  = __builtin_shufflevector(lo, hi8, 0, 1, 2, 3, 4, 5, 6, 7,
                                      8, 9, 10, 11, 12, 13, 14, 15);
#pragma unroll
    for (int nh = 0; nh < 4; ++nh) {
      v16h bv = *(const v16h*)(wfrag + ((c * 4 + nh) * 32 + lane) * 16);
      acc[nh] = __builtin_amdgcn_wmma_f32_16x16x32_f16(false, a, false, bv, (short)0, acc[nh], false, false);
    }
  }
#pragma unroll
  for (int nh = 0; nh < 4; ++nh) {
    float bias = fb1[nh * 16 + nl];
#pragma unroll
    for (int r = 0; r < 8; ++r) {
      int m = tile * 16 + r + 8 * h;
      float v = acc[nh][r] + bias;
      v = v > 0.0f ? v : 0.0f;
      h3[(long)m * 64 + nh * 16 + nl] = v;
    }
  }
}

// ---------------- fc2 + batch statistics ----------------
__global__ void k_zero(float* __restrict__ acc) {
  if (threadIdx.x < 8) acc[threadIdx.x] = 0.0f;
}

__global__ void __launch_bounds__(256) k_fc2(const float* __restrict__ h3,
                                             const float* __restrict__ fw2,
                                             const float* __restrict__ fb2,
                                             float* __restrict__ o,
                                             float* __restrict__ accum) {
  __shared__ float sw[256];                            // fw2 is exactly 4x64
  int tid = threadIdx.x;
  sw[tid] = fw2[tid];
  __syncthreads();
  int b = blockIdx.x * 256 + tid;                      // 8192 exact
  const float* hr = h3 + (long)b * 64;
  float ov[4];
#pragma unroll
  for (int j = 0; j < 4; ++j) {
    float s = fb2[j];
#pragma unroll
    for (int k = 0; k < 64; ++k) s += hr[k] * sw[j * 64 + k];
    ov[j] = s;
    o[(long)b * 4 + j] = s;
  }
#pragma unroll
  for (int j = 0; j < 4; ++j) {
    float s = ov[j], ss = ov[j] * ov[j];
    for (int off = 16; off > 0; off >>= 1) {
      s  += __shfl_xor(s,  off, 32);
      ss += __shfl_xor(ss, off, 32);
    }
    if ((tid & 31) == 0) {
      atomicAdd(accum + j, s);
      atomicAdd(accum + 4 + j, ss);
    }
  }
}

__global__ void k_bn(const float* __restrict__ accum, const float* __restrict__ gamma,
                     const float* __restrict__ beta, float* __restrict__ ss) {
  int j = threadIdx.x;
  if (j < 4) {
    float mu  = accum[j] * (1.0f / (float)BATCH);
    float var = accum[4 + j] * (1.0f / (float)BATCH) - mu * mu;
    float inv = rsqrtf(var + 1e-5f);
    float g = gamma[j] * inv;
    ss[j]     = g;
    ss[4 + j] = beta[j] - mu * g;
  }
}

__global__ void k_norm(const float* __restrict__ o, const float* __restrict__ ss,
                       float* __restrict__ out, int total) {
  int t = blockIdx.x * blockDim.x + threadIdx.x;
  if (t >= total) return;
  int j = t & 3;
  out[t] = o[t] * ss[j] + ss[4 + j];
}

extern "C" void kernel_launch(void* const* d_in, const int* in_sizes, int n_in,
                              void* d_out, int out_size, void* d_ws, size_t ws_size,
                              hipStream_t stream) {
  const float* x     = (const float*)d_in[0];
  const float* w1    = (const float*)d_in[1];
  const float* b1    = (const float*)d_in[2];
  const float* w2    = (const float*)d_in[3];
  const float* b2    = (const float*)d_in[4];
  const float* fw1   = (const float*)d_in[5];
  const float* fb1   = (const float*)d_in[6];
  const float* fw2   = (const float*)d_in[7];
  const float* fb2   = (const float*)d_in[8];
  const float* gamma = (const float*)d_in[9];
  const float* beta  = (const float*)d_in[10];
  float* out = (float*)d_out;

  char* ws = (char*)d_ws;
  _Float16* h1     = (_Float16*)(ws + OFF_H1);
  float*    q      = (float*)   (ws + OFF_Q);
  _Float16* c2     = (_Float16*)(ws + OFF_C2);
  _Float16* h2     = (_Float16*)(ws + OFF_H2);
  float*    h3     = (float*)   (ws + OFF_H3);
  float*    o      = (float*)   (ws + OFF_O);
  float*    accum  = (float*)   (ws + OFF_ACC);
  float*    ss     = (float*)   (ws + OFF_SS);
  _Float16* wbfrag = (_Float16*)(ws + OFF_WB);
  _Float16* fwfrag = (_Float16*)(ws + OFF_FW);

  (void)in_sizes; (void)n_in; (void)out_size; (void)ws_size;

  // weight pre-swizzle (independent of activations, launch first)
  k_prepw2 <<<1, 320, 0, stream>>>(w2, wbfrag);
  k_prepfw1<<<25, 256, 0, stream>>>(fw1, fwfrag);

  // preprocess: 8192*28*28 outputs
  {
    int total = BATCH * 784;
    k_pre<<<(total + 255) / 256, 256, 0, stream>>>(x, q, total);
  }
  // conv1 + relu + pool: 8192*16*14*14 outputs
  {
    int total = BATCH * 16 * 196;
    k_conv1<<<(total + 255) / 256, 256, 0, stream>>>(q, w1, b1, h1, total);
  }
  // conv2 via WMMA: 100,352 M-tiles, 8 waves/block
  k_conv2<<<100352 / 8, 256, 0, stream>>>(h1, wbfrag, b2, c2);
  // pool + flatten: 8192*32*7*7 outputs
  {
    int total = BATCH * 32 * 49;
    k_pool<<<(total + 255) / 256, 256, 0, stream>>>(c2, h2, total);
  }
  // fc1 via WMMA: 512 M-tiles, 8 waves/block
  k_fc1<<<512 / 8, 256, 0, stream>>>(h2, fwfrag, fb1, h3);
  // fc2 + batch stats + batchnorm
  k_zero<<<1, 32, 0, stream>>>(accum);
  k_fc2<<<BATCH / 256, 256, 0, stream>>>(h3, fw2, fb2, o, accum);
  k_bn<<<1, 32, 0, stream>>>(accum, gamma, beta, ss);
  {
    int total = BATCH * 4;
    k_norm<<<(total + 255) / 256, 256, 0, stream>>>(o, ss, out, total);
  }
}
